// PhiEngram_14542759264782
// MI455X (gfx1250) — matrix operand order repaired and checked by reading, not compiled, not run on hardware
//
#include <hip/hip_runtime.h>

// ---------------------------------------------------------------------------
// PhiEngram block for MI455X (gfx1250, wave32).
// GEMMs (103 GFLOP) run on v_wmma_f32_16x16x32_bf16 with fp32 accumulation.
// Block tiles are staged into LDS by the Tensor Data Mover (TENSORcnt +
// s_wait_tensorcnt), removing the 4x/2x intra-block L2 re-fetch of the
// direct-global scheme. All other stages are bandwidth-bound and fused.
// ---------------------------------------------------------------------------

#define BB   4
#define LL   2048
#define GG   2
#define DD   2048
#define HH   16
#define DE   64
#define EE   1024      // H * De
#define NNT  32768
#define GD   4096      // G * D
#define BL   8192      // B * L
#define DIL  3
#define PAD  9         // (K-1)*DIL
#define BK   64        // K-block staged in LDS per TDM transfer

typedef __bf16 bf16_t;
typedef __attribute__((ext_vector_type(16))) __bf16 v16bf;
typedef __attribute__((ext_vector_type(8)))  __bf16 v8bf;
typedef __attribute__((ext_vector_type(8)))  float  v8f;
typedef __attribute__((ext_vector_type(4)))  unsigned int v4u;
typedef __attribute__((ext_vector_type(8)))  int v8i;
typedef __attribute__((ext_vector_type(4)))  int v4i;

__device__ __forceinline__ bf16_t f2bf(float f) {
  union { float f; unsigned u; } in; in.f = f;
  unsigned u = in.u;
  u += 0x7FFFu + ((u >> 16) & 1u);        // round-to-nearest-even
  union { unsigned short s; bf16_t b; } out;
  out.s = (unsigned short)(u >> 16);
  return out.b;
}

// ---------------------------------------------------------------------------
// 1) Gather hashed embeddings and cast to bf16.
// ---------------------------------------------------------------------------
__global__ __launch_bounds__(256) void gather_cast_kernel(
    const int* __restrict__ ids, const float* __restrict__ table,
    bf16_t* __restrict__ embb) {
  size_t t = (size_t)blockIdx.x * 256 + threadIdx.x;   // over BL*EE
  int    e  = (int)(t & (EE - 1));
  size_t bl = t >> 10;
  int h = e >> 6;
  int j = e & 63;
  int id = ids[bl * HH + h];
  float v = table[((size_t)h * NNT + (size_t)id) * DE + j];
  embb[t] = f2bf(v);
}

// ---------------------------------------------------------------------------
// 2) Cast weight matrices to bf16.
// ---------------------------------------------------------------------------
__global__ __launch_bounds__(256) void cast_bf16_kernel(
    const float* __restrict__ src, bf16_t* __restrict__ dst, int n) {
  int t = blockIdx.x * 256 + threadIdx.x;
  if (t < n) dst[t] = f2bf(src[t]);
}

// ---------------------------------------------------------------------------
// TDM 2-D tile load: rows x BK bf16 elements (row stride = stride_elts) from
// global into LDS at lds_off.  D# layout per CDNA5 ISA 8.3/8.4:
//   g0: count=1 | lds_addr | global_addr[56:0] | type=2
//   g1: data_size=2B, tensor_dim0/1, tile_dim0=BK, tile_dim1=rows,
//       tensor_dim0_stride=stride_elts
// This toolchain exposes the 6-arg builtin form (clang-23 / therock-10.0):
//   (uint32x4 g0, int32x8 g1, int32x4, int32x4, int32x8, i32 cpol)
// ---------------------------------------------------------------------------
#if __has_builtin(__builtin_amdgcn_tensor_load_to_lds)
#define HAVE_TDM 1
__device__ __forceinline__ void tdm_load_2d(unsigned lds_off, const void* gptr,
                                            unsigned rows, unsigned tensor_d1,
                                            unsigned stride_elts) {
  unsigned long long ga = (unsigned long long)gptr;
  v4u g0;
  g0[0] = 1u;                                         // count=1, user mode
  g0[1] = lds_off;                                    // lds_addr (bytes)
  g0[2] = (unsigned)(ga & 0xffffffffu);               // global_addr lo
  g0[3] = (unsigned)((ga >> 32) & 0x01ffffffu)        // global_addr hi
        | (2u << 30);                                 // type = 2 ("image")
  v8i g1;
  g1[0] = 0x00010000;                                 // data_size = 1 -> 2B
  g1[1] = (int)((EE & 0xffffu) << 16);                // tensor_dim0 lo16
  g1[2] = (int)(((EE >> 16) & 0xffffu)                // tensor_dim0 hi16
        | ((tensor_d1 & 0xffffu) << 16));             // tensor_dim1 lo16
  g1[3] = (int)(((tensor_d1 >> 16) & 0xffffu)         // tensor_dim1 hi16
        | ((BK & 0xffffu) << 16));                    // tile_dim0 = BK
  g1[4] = (int)(rows & 0xffffu);                      // tile_dim1; tile_dim2=0
  g1[5] = (int)stride_elts;                           // tensor_dim0_stride lo
  g1[6] = 0;                                          // stride hi / dim1_stride
  g1[7] = 0;
  v4i z4 = {0, 0, 0, 0};
  v8i z8 = {0, 0, 0, 0, 0, 0, 0, 0};
  __builtin_amdgcn_tensor_load_to_lds(g0, g1, z4, z4, z8, 0);
}
#else
#define HAVE_TDM 0
#endif

// ---------------------------------------------------------------------------
// 3) C[m,n] = sum_k A[m,k]*W[n,k] + bias[n], K = EE = 1024.
//    Block tile 64x128 (8 waves, each 32x32 = 2x2 WMMA accumulators).
//    TDM stages A(64xBK) and B(128xBK) into LDS; fragments come from LDS.
//    Row/col per CDNA5 ISA 7.12.2; k order here is contiguous k = 16*lhalf + s in A and B.
// ---------------------------------------------------------------------------
__global__ __launch_bounds__(256) void gemm_bf16_kernel(
    const bf16_t* __restrict__ A,   // (BL x EE) row-major
    const bf16_t* __restrict__ W,   // (Nn x EE) row-major (acts as B^T)
    const float*  __restrict__ bias,
    float* __restrict__ C, int Nn) {
  const int tid   = threadIdx.x;
  const int wave  = tid >> 5;
  const int lane  = tid & 31;
  const int lhalf = lane >> 4;     // 0 | 1
  const int lrow  = lane & 15;
  const int rowL  = (wave >> 2) * 32;          // wave tile row within block
  const int colL  = (wave & 3)  * 32;          // wave tile col within block
  const int row0  = blockIdx.y * 64  + rowL;
  const int col0  = blockIdx.x * 128 + colL;

  v8f acc[2][2] = {};

#if HAVE_TDM
  __shared__ __align__(16) bf16_t As[64  * BK];
  __shared__ __align__(16) bf16_t Bs[128 * BK];
  const unsigned ldsA = (unsigned)(unsigned long long)(As);
  const unsigned ldsB = (unsigned)(unsigned long long)(Bs);
  const bf16_t* gA = A + (size_t)(blockIdx.y * 64)  * EE;
  const bf16_t* gB = W + (size_t)(blockIdx.x * 128) * EE;

  for (int k0 = 0; k0 < EE; k0 += BK) {
    if (wave == 0) {
      tdm_load_2d(ldsA, gA + k0, 64,  BL, EE);
      tdm_load_2d(ldsB, gB + k0, 128, (unsigned)Nn, EE);
      __builtin_amdgcn_s_wait_tensorcnt(0);
    }
    __syncthreads();                           // LDS tiles ready

    #pragma unroll
    for (int kk = 0; kk < BK; kk += 32) {
      union { v16bf v; v8bf h[2]; } a0, a1;
      const bf16_t* ar0 = As + (size_t)(rowL + lrow)      * BK + kk;
      const bf16_t* ar1 = As + (size_t)(rowL + 16 + lrow) * BK + kk;
      a0.h[0] = *(const v8bf*)(ar0 + lhalf * 16);
      a0.h[1] = *(const v8bf*)(ar0 + 8 + lhalf * 16);
      a1.h[0] = *(const v8bf*)(ar1 + lhalf * 16);
      a1.h[1] = *(const v8bf*)(ar1 + 8 + lhalf * 16);
      v16bf b0 = *(const v16bf*)(Bs + (size_t)(colL + lrow)      * BK + kk + lhalf * 16);
      v16bf b1 = *(const v16bf*)(Bs + (size_t)(colL + 16 + lrow) * BK + kk + lhalf * 16);

      acc[0][0] = __builtin_amdgcn_wmma_f32_16x16x32_bf16(
          false, a0.v, false, b0, (short)0, acc[0][0], false, false);
      acc[0][1] = __builtin_amdgcn_wmma_f32_16x16x32_bf16(
          false, a0.v, false, b1, (short)0, acc[0][1], false, false);
      acc[1][0] = __builtin_amdgcn_wmma_f32_16x16x32_bf16(
          false, a1.v, false, b0, (short)0, acc[1][0], false, false);
      acc[1][1] = __builtin_amdgcn_wmma_f32_16x16x32_bf16(
          false, a1.v, false, b1, (short)0, acc[1][1], false, false);
    }
    __syncthreads();                           // done reading before next TDM
  }
#else
  // Fallback: direct-global fragments.
  const bf16_t* arow0 = A + (size_t)(row0 + lrow)      * EE;
  const bf16_t* arow1 = A + (size_t)(row0 + 16 + lrow) * EE;
  const bf16_t* brow0 = W + (size_t)(col0 + lrow)      * EE;
  const bf16_t* brow1 = W + (size_t)(col0 + 16 + lrow) * EE;
  for (int k = 0; k < EE; k += 32) {
    __builtin_prefetch(arow0 + k + 32, 0, 1);
    __builtin_prefetch(brow0 + k + 32, 0, 1);
    union { v16bf v; v8bf h[2]; } a0, a1;
    a0.h[0] = *(const v8bf*)(arow0 + k + lhalf * 16);
    a0.h[1] = *(const v8bf*)(arow0 + k + 8 + lhalf * 16);
    a1.h[0] = *(const v8bf*)(arow1 + k + lhalf * 16);
    a1.h[1] = *(const v8bf*)(arow1 + k + 8 + lhalf * 16);
    v16bf b0 = *(const v16bf*)(brow0 + k + lhalf * 16);
    v16bf b1 = *(const v16bf*)(brow1 + k + lhalf * 16);
    acc[0][0] = __builtin_amdgcn_wmma_f32_16x16x32_bf16(
        false, a0.v, false, b0, (short)0, acc[0][0], false, false);
    acc[0][1] = __builtin_amdgcn_wmma_f32_16x16x32_bf16(
        false, a0.v, false, b1, (short)0, acc[0][1], false, false);
    acc[1][0] = __builtin_amdgcn_wmma_f32_16x16x32_bf16(
        false, a1.v, false, b0, (short)0, acc[1][0], false, false);
    acc[1][1] = __builtin_amdgcn_wmma_f32_16x16x32_bf16(
        false, a1.v, false, b1, (short)0, acc[1][1], false, false);
  }
#endif

  // C/D layout: element v -> row = lhalf*8 + v (within 16), col = lrow.
  #pragma unroll
  for (int i = 0; i < 2; ++i) {
    #pragma unroll
    for (int j = 0; j < 2; ++j) {
      int   ccol = col0 + j * 16 + lrow;
      float bv   = bias[ccol];
      #pragma unroll
      for (int v = 0; v < 8; ++v) {
        int r = row0 + i * 16 + lhalf * 8 + v;
        C[(size_t)r * Nn + ccol] = acc[i][j][v] + bv;
      }
    }
  }
}

// ---------------------------------------------------------------------------
// 4) Fused gate + x_norm: one block per (b,l,g).
// ---------------------------------------------------------------------------
__global__ __launch_bounds__(256) void gate_xnorm_kernel(
    const float* __restrict__ keyp, const float* __restrict__ hid,
    const float* __restrict__ vproj,
    const float* __restrict__ n1w,  const float* __restrict__ n2w,
    const float* __restrict__ scw,
    float* __restrict__ gateb, float* __restrict__ xnorm) {
  const int    tid = threadIdx.x;
  const int    g   = blockIdx.x & (GG - 1);
  const size_t bl  = blockIdx.x >> 1;

  const float* krow = keyp  + bl * GD + (size_t)g * DD;
  const float* hrow = hid   + (bl * GG + g) * (size_t)DD;
  const float* vrow = vproj + bl * DD;
  const float* w1   = n1w + g * DD;
  const float* w2   = n2w + g * DD;
  const float* wsc  = scw + g * DD;

  float pK = 0.f, pQ = 0.f, pKQ = 0.f, pV = 0.f;
  for (int d = tid; d < DD; d += 256) {
    float kv = krow[d], hv = hrow[d], vv = vrow[d];
    pK  += kv * kv;
    pQ  += hv * hv;
    pKQ += (kv * w1[d]) * (hv * w2[d]);
    pV  += vv * vv;
  }

  __shared__ float sK[256], sQ[256], sKQ[256], sV[256];
  sK[tid] = pK; sQ[tid] = pQ; sKQ[tid] = pKQ; sV[tid] = pV;
  __syncthreads();
  for (int s = 128; s > 0; s >>= 1) {
    if (tid < s) {
      sK[tid]  += sK[tid + s];  sQ[tid] += sQ[tid + s];
      sKQ[tid] += sKQ[tid + s]; sV[tid] += sV[tid + s];
    }
    __syncthreads();
  }

  __shared__ float bGate, bRs;
  if (tid == 0) {
    const float EPSD = 1.1920928955078125e-07f;   // float32 eps
    float rsK = rsqrtf(sK[0] * (1.0f / DD) + EPSD);
    float rsQ = rsqrtf(sQ[0] * (1.0f / DD) + EPSD);
    float gg  = rsK * rsQ * sKQ[0] * (1.0f / sqrtf((float)DD));
    float t   = sqrtf(fmaxf(fabsf(gg), 1e-6f));
    float gs  = (gg > 0.f) ? t : ((gg < 0.f) ? -t : 0.f);
    float gate = 1.f / (1.f + __expf(-gs));
    gateb[bl * GG + g] = gate;
    bGate = gate;
    bRs   = rsqrtf(gate * gate * (sV[0] * (1.0f / DD)) + 1e-5f);
  }
  __syncthreads();

  float gate = bGate, rs = bRs;
  for (int d = tid; d < DD; d += 256) {
    float val = gate * vrow[d];
    xnorm[bl * GD + (size_t)g * DD + d] = val * rs * wsc[d];
  }
}

// ---------------------------------------------------------------------------
// 5) Depthwise causal dilated conv (K=4, DIL=3) + SiLU + residual + cache.
// ---------------------------------------------------------------------------
__global__ __launch_bounds__(256) void conv_out_kernel(
    const float* __restrict__ xnorm, const float* __restrict__ vproj,
    const float* __restrict__ gateb, const float* __restrict__ convw,
    float* __restrict__ outp, float* __restrict__ cache) {
  const int tid  = threadIdx.x;
  int bidx = blockIdx.x;
  int cblk = bidx & 15;            // 16 blocks of 256 channels
  int r    = (bidx >> 4) % 3;      // residue class mod DIL
  int b    = bidx / 48;
  int c    = cblk * 256 + tid;     // channel = g*D + d
  int g    = c >> 11;
  int dd   = c & (DD - 1);

  float w0 = convw[c * 4 + 0], w1 = convw[c * 4 + 1];
  float w2 = convw[c * 4 + 2], w3 = convw[c * 4 + 3];

  float xm9 = 0.f, xm6 = 0.f, xm3 = 0.f;   // taps l-9, l-6, l-3 (zero-pad)
  for (int l = r; l < LL; l += DIL) {
    size_t base = (size_t)b * LL + l;
    float xc = xnorm[base * GD + c];
    float y  = w0 * xm9 + w1 * xm6 + w2 * xm3 + w3 * xc;
    float sy = y / (1.f + __expf(-y));                 // silu
    float gate = gateb[base * GG + g];
    float val  = gate * vproj[base * DD + dd];
    outp[base * GD + c] = val + sy;
    if (l >= LL - PAD)
      cache[((size_t)b * GD + c) * PAD + (l - (LL - PAD))] = xc;
    xm9 = xm6; xm6 = xm3; xm3 = xc;
  }
}

// ---------------------------------------------------------------------------
extern "C" void kernel_launch(void* const* d_in, const int* in_sizes, int n_in,
                              void* d_out, int out_size, void* d_ws, size_t ws_size,
                              hipStream_t stream) {
  (void)in_sizes; (void)n_in; (void)out_size; (void)ws_size;

  const float* hid   = (const float*)d_in[0];
  const int*   ids   = (const int*)  d_in[1];
  const float* table = (const float*)d_in[2];
  const float* Wk    = (const float*)d_in[3];
  const float* bk    = (const float*)d_in[4];
  const float* Wv    = (const float*)d_in[5];
  const float* bv    = (const float*)d_in[6];
  const float* n1w   = (const float*)d_in[7];
  const float* n2w   = (const float*)d_in[8];
  const float* cw    = (const float*)d_in[9];
  const float* scw   = (const float*)d_in[10];

  float* outp  = (float*)d_out;
  float* cache = outp + (size_t)BL * GD;    // new_cache after output

  // Workspace carve-out (256B aligned): ~350 MB total.
  char*  ws  = (char*)d_ws;
  size_t off = 0;
  auto carve = [&](size_t bytes) -> void* {
    void* p = ws + off;
    off = (off + bytes + 255) & ~(size_t)255;
    return p;
  };
  bf16_t* embb  = (bf16_t*)carve((size_t)BL * EE * 2);
  bf16_t* Wkb   = (bf16_t*)carve((size_t)GD * EE * 2);
  bf16_t* Wvb   = (bf16_t*)carve((size_t)DD * EE * 2);
  float*  keyp  = (float*) carve((size_t)BL * GD * 4);
  float*  vproj = (float*) carve((size_t)BL * DD * 4);
  float*  gateb = (float*) carve((size_t)BL * GG * 4);
  float*  xnorm = (float*) carve((size_t)BL * GD * 4);

  gather_cast_kernel<<<(BL * EE) / 256, 256, 0, stream>>>(ids, table, embb);
  cast_bf16_kernel<<<(GD * EE) / 256, 256, 0, stream>>>(Wk, Wkb, GD * EE);
  cast_bf16_kernel<<<(DD * EE) / 256, 256, 0, stream>>>(Wv, Wvb, DD * EE);

  // keyp = emb * Wk^T + bk   (M=8192, N=4096, K=1024)
  gemm_bf16_kernel<<<dim3(GD / 128, BL / 64), 256, 0, stream>>>(
      embb, Wkb, bk, keyp, GD);
  // vproj = emb * Wv^T + bv  (M=8192, N=2048, K=1024)
  gemm_bf16_kernel<<<dim3(DD / 128, BL / 64), 256, 0, stream>>>(
      embb, Wvb, bv, vproj, DD);

  gate_xnorm_kernel<<<BL * GG, 256, 0, stream>>>(
      keyp, hid, vproj, n1w, n2w, scw, gateb, xnorm);

  conv_out_kernel<<<BB * 3 * (GD / 256), 256, 0, stream>>>(
      xnorm, vproj, gateb, cw, outp, cache);
}
